// Deconv_DeformBlock_74302934221276
// MI455X (gfx1250) — compile-verified, matrix operand order checked by reading, not run on hardware
//
#include <hip/hip_runtime.h>

typedef __attribute__((ext_vector_type(16))) _Float16 v16h;
typedef __attribute__((ext_vector_type(8)))  _Float16 v8h;
typedef __attribute__((ext_vector_type(8)))  float    v8f;

#define HW   65536   // 256*256
#define WIMG 256

static_assert(sizeof(v16h) == 32, "v16h must be 32B");
static_assert(sizeof(v8h)  == 16, "v8h must be 16B");

union AB { v16h v; v8h h2[2]; };

__device__ __forceinline__ v8h splat8(_Float16 s) {
    v8h v = {s, s, s, s, s, s, s, s};
    return v;
}

// ---------------------------------------------------------------------------
// Pack x: NCHW f32 -> NHWC f16.  xh[(b*HW + y*256 + x)*64 + c]
// ---------------------------------------------------------------------------
__global__ __launch_bounds__(256) void pack_x(const float* __restrict__ xin,
                                              _Float16* __restrict__ xh) {
    int tid = blockIdx.x * 256 + threadIdx.x;   // 2*65536*64 threads
    int c  = tid & 63;
    int p  = tid >> 6;            // b*HW + spatial
    int b  = p >> 16;
    int sp = p & 65535;
    xh[tid] = (_Float16)xin[((size_t)(b * 64 + c) << 16) + sp];
}

// ---------------------------------------------------------------------------
// Weight prep: emit f16 weights ALREADY in wave32 WMMA B-fragment order:
//   Wf[((q*NT + n)*32 + lane)*16 + half]
// half<8 -> K = g*8+half, half>=8 -> K = 16+g*8+(half-8); g=lane>>4,
// col=lane&15, kk = q*32+K, t = kk>>6 (tap), c = kk&63 (channel), o=n*16+col.
//  W1f (NT=4): conv1 = w_deconv[c][o][8-t]   (flip + transpose)
//  W2f (NT=2): offset conv = w_off[j][c][t]  (j>=18 zero-padded)
//  W3f (NT=4): deform conv = w_def[o][c][t]
// ---------------------------------------------------------------------------
__global__ __launch_bounds__(256) void prep_weights(const float* __restrict__ wdc,
                                                    const float* __restrict__ woff,
                                                    const float* __restrict__ wdef,
                                                    _Float16* __restrict__ W1f,
                                                    _Float16* __restrict__ W2f,
                                                    _Float16* __restrict__ W3f) {
    int tid = blockIdx.x * 256 + threadIdx.x;
    if (tid < 18 * 4 * 32 * 16) {
        int half = tid & 15;
        int lane = (tid >> 4) & 31;
        int rest = tid >> 9;                // 0..71
        int n = rest & 3, q = rest >> 2;
        int g = lane >> 4, col = lane & 15;
        int k = (half < 8) ? ((g << 3) + half) : (16 + (g << 3) + (half & 7));
        int kk = (q << 5) + k;
        int t = kk >> 6, c = kk & 63;
        int o = (n << 4) + col;
        W1f[tid] = (_Float16)wdc[(c * 64 + o) * 9 + (8 - t)];
        W3f[tid] = (_Float16)wdef[(o * 64 + c) * 9 + t];
    }
    if (tid < 18 * 2 * 32 * 16) {
        int half = tid & 15;
        int lane = (tid >> 4) & 31;
        int rest = tid >> 9;                // 0..35
        int n = rest & 1, q = rest >> 1;
        int g = lane >> 4, col = lane & 15;
        int k = (half < 8) ? ((g << 3) + half) : (16 + (g << 3) + (half & 7));
        int kk = (q << 5) + k;
        int t = kk >> 6, c = kk & 63;
        int j = (n << 4) + col;
        W2f[tid] = (j < 18) ? (_Float16)woff[(j * 64 + c) * 9 + t] : (_Float16)0.0f;
    }
}

// ---------------------------------------------------------------------------
// Stage 1: h = relu(conv3x3(xh) + b_deconv)  (64 -> 64, pad 1), h NHWC f16.
// M=32 tiles: each wave computes 32 pixels x 64 channels; every B fragment
// feeds two WMMAs. 4096 waves -> 512 blocks.
// ---------------------------------------------------------------------------
__global__ __launch_bounds__(256) void conv1_wmma(const _Float16* __restrict__ xh,
                                                  const v16h* __restrict__ W1f,
                                                  const float* __restrict__ bias,
                                                  _Float16* __restrict__ hout) {
    int lane = threadIdx.x & 31;
    int g = lane >> 4;
    int lc = lane & 15;
    int wg = blockIdx.x * 8 + (threadIdx.x >> 5);
    int b = wg >> 11;
    int rem = wg & 2047;
    int y = rem >> 3;
    int xBase = (rem & 7) << 5;
    int xPix0 = xBase + lc;

    const _Float16* xb = xh + ((size_t)b << 16) * 64;
    v8f acc[2][4] = {{v8f{}, v8f{}, v8f{}, v8f{}}, {v8f{}, v8f{}, v8f{}, v8f{}}};

#pragma unroll
    for (int t = 0; t < 9; ++t) {
        int sy = y + t / 3 - 1;
        int sx0 = xPix0 + t % 3 - 1;
        int sx1 = sx0 + 16;
        bool ok0 = ((unsigned)sy < 256u) && ((unsigned)sx0 < 256u);
        bool ok1 = ((unsigned)sy < 256u) && ((unsigned)sx1 < 256u);
        const _Float16* p0 = xb + ((size_t)(((sy & 255) << 8) + (sx0 & 255)) << 6);
        const _Float16* p1 = xb + ((size_t)(((sy & 255) << 8) + (sx1 & 255)) << 6);
#pragma unroll
        for (int half = 0; half < 2; ++half) {
            int c0 = (half << 5) + (g << 3);
            AB a0, a1;
            a0.h2[0] = *(const v8h*)(p0 + c0);
            a0.h2[1] = *(const v8h*)(p0 + c0 + 16);
            a1.h2[0] = *(const v8h*)(p1 + c0);
            a1.h2[1] = *(const v8h*)(p1 + c0 + 16);
            if (!ok0) { a0.h2[0] = v8h{}; a0.h2[1] = v8h{}; }
            if (!ok1) { a1.h2[0] = v8h{}; a1.h2[1] = v8h{}; }
            int q = (t << 1) + half;
#pragma unroll
            for (int n = 0; n < 4; ++n) {
                v16h bv = W1f[((q << 2) + n) * 32 + lane];
                acc[0][n] = __builtin_amdgcn_wmma_f32_16x16x32_f16(
                    false, a0.v, false, bv, (short)0, acc[0][n], false, false);
                acc[1][n] = __builtin_amdgcn_wmma_f32_16x16x32_f16(
                    false, a1.v, false, bv, (short)0, acc[1][n], false, false);
            }
        }
        if (t < 8) {  // prefetch next tap's activation row (global_prefetch_b8)
            int t2 = t + 1;
            int sy2 = y + t2 / 3 - 1;
            int sx2 = xPix0 + t2 % 3 - 1;
            __builtin_prefetch(xb + ((size_t)(((sy2 & 255) << 8) + (sx2 & 255)) << 6), 0, 3);
        }
    }
#pragma unroll
    for (int m = 0; m < 2; ++m) {
#pragma unroll
        for (int n = 0; n < 4; ++n) {
            int o = (n << 4) + lc;
            float bo = bias[o];
            _Float16* hp = hout +
                ((size_t)((b << 16) + (y << 8) + xBase + (m << 4) + (g << 3)) << 6) + o;
#pragma unroll
            for (int i = 0; i < 8; ++i)
                hp[(size_t)i << 6] = (_Float16)fmaxf(acc[m][n][i] + bo, 0.0f);
        }
    }
}

// ---------------------------------------------------------------------------
// Stage 2: offset = conv3x3(h) + b_off   (64 -> 18, N padded to 32), M=32.
// offs layout: [b][pixel][18] f32  (dy/dx pairs adjacent for deform stage)
// ---------------------------------------------------------------------------
__global__ __launch_bounds__(256) void conv2_wmma(const _Float16* __restrict__ h,
                                                  const v16h* __restrict__ W2f,
                                                  const float* __restrict__ boff,
                                                  float* __restrict__ offs) {
    int lane = threadIdx.x & 31;
    int g = lane >> 4;
    int lc = lane & 15;
    int wg = blockIdx.x * 8 + (threadIdx.x >> 5);
    int b = wg >> 11;
    int rem = wg & 2047;
    int y = rem >> 3;
    int xBase = (rem & 7) << 5;
    int xPix0 = xBase + lc;

    const _Float16* hb = h + ((size_t)b << 16) * 64;
    v8f acc[2][2] = {{v8f{}, v8f{}}, {v8f{}, v8f{}}};

#pragma unroll
    for (int t = 0; t < 9; ++t) {
        int sy = y + t / 3 - 1;
        int sx0 = xPix0 + t % 3 - 1;
        int sx1 = sx0 + 16;
        bool ok0 = ((unsigned)sy < 256u) && ((unsigned)sx0 < 256u);
        bool ok1 = ((unsigned)sy < 256u) && ((unsigned)sx1 < 256u);
        const _Float16* p0 = hb + ((size_t)(((sy & 255) << 8) + (sx0 & 255)) << 6);
        const _Float16* p1 = hb + ((size_t)(((sy & 255) << 8) + (sx1 & 255)) << 6);
#pragma unroll
        for (int half = 0; half < 2; ++half) {
            int c0 = (half << 5) + (g << 3);
            AB a0, a1;
            a0.h2[0] = *(const v8h*)(p0 + c0);
            a0.h2[1] = *(const v8h*)(p0 + c0 + 16);
            a1.h2[0] = *(const v8h*)(p1 + c0);
            a1.h2[1] = *(const v8h*)(p1 + c0 + 16);
            if (!ok0) { a0.h2[0] = v8h{}; a0.h2[1] = v8h{}; }
            if (!ok1) { a1.h2[0] = v8h{}; a1.h2[1] = v8h{}; }
            int q = (t << 1) + half;
#pragma unroll
            for (int n = 0; n < 2; ++n) {
                v16h bv = W2f[((q << 1) + n) * 32 + lane];
                acc[0][n] = __builtin_amdgcn_wmma_f32_16x16x32_f16(
                    false, a0.v, false, bv, (short)0, acc[0][n], false, false);
                acc[1][n] = __builtin_amdgcn_wmma_f32_16x16x32_f16(
                    false, a1.v, false, bv, (short)0, acc[1][n], false, false);
            }
        }
        if (t < 8) {
            int t2 = t + 1;
            int sy2 = y + t2 / 3 - 1;
            int sx2 = xPix0 + t2 % 3 - 1;
            __builtin_prefetch(hb + ((size_t)(((sy2 & 255) << 8) + (sx2 & 255)) << 6), 0, 3);
        }
    }
#pragma unroll
    for (int m = 0; m < 2; ++m) {
#pragma unroll
        for (int n = 0; n < 2; ++n) {
            int j = (n << 4) + lc;
            if (j < 18) {
                float bo = boff[j];
                float* op = offs +
                    (size_t)((b << 16) + (y << 8) + xBase + (m << 4) + (g << 3)) * 18 + j;
#pragma unroll
                for (int i = 0; i < 8; ++i)
                    op[(size_t)i * 18] = acc[m][n][i] + bo;
            }
        }
    }
}

// ---------------------------------------------------------------------------
// Stage 3: deformable conv (M=16). Bilinear corner addresses / weights are
// computed once per lane per tap; the 4-corner blend runs in packed f16
// (v_pk_fma_f16) on whole v8h vectors. out = relu(GEMM + b_def), f32 NCHW.
// ---------------------------------------------------------------------------
__global__ __launch_bounds__(256) void deform_wmma(const _Float16* __restrict__ h,
                                                   const float* __restrict__ offs,
                                                   const v16h* __restrict__ W3f,
                                                   const float* __restrict__ bdef,
                                                   float* __restrict__ out) {
    int lane = threadIdx.x & 31;
    int g = lane >> 4;
    int lc = lane & 15;
    int wg = blockIdx.x * 8 + (threadIdx.x >> 5);
    int b = wg >> 12;
    int rem = wg & 4095;
    int y = rem >> 4;
    int xBase = (rem & 15) << 4;
    int xPix = xBase + lc;

    const _Float16* hb = h + ((size_t)b << 16) * 64;
    const float* offp = offs + (size_t)((b << 16) + (y << 8) + xPix) * 18;
    v8f acc[4] = {v8f{}, v8f{}, v8f{}, v8f{}};

#pragma unroll
    for (int t = 0; t < 9; ++t) {
        float2 d = *(const float2*)(offp + (t << 1));   // (dy, dx), 8B aligned
        float py = (float)(y - 1 + t / 3) + d.x;
        float px = (float)(xPix - 1 + t % 3) + d.y;
        float fy = floorf(py), fx = floorf(px);
        int iy0 = (int)fy, ix0 = (int)fx;
        int iy1 = iy0 + 1, ix1 = ix0 + 1;
        float wy1 = py - fy, wx1 = px - fx;
        float wy0 = 1.0f - wy1, wx0 = 1.0f - wx1;
        float vy0 = ((unsigned)iy0 < 256u) ? 1.f : 0.f;
        float vy1 = ((unsigned)iy1 < 256u) ? 1.f : 0.f;
        float vx0 = ((unsigned)ix0 < 256u) ? 1.f : 0.f;
        float vx1 = ((unsigned)ix1 < 256u) ? 1.f : 0.f;
        v8h W00 = splat8((_Float16)(wy0 * wx0 * vy0 * vx0));
        v8h W01 = splat8((_Float16)(wy0 * wx1 * vy0 * vx1));
        v8h W10 = splat8((_Float16)(wy1 * wx0 * vy1 * vx0));
        v8h W11 = splat8((_Float16)(wy1 * wx1 * vy1 * vx1));
        const _Float16* hc00 = hb + ((size_t)(((iy0 & 255) << 8) + (ix0 & 255)) << 6);
        const _Float16* hc01 = hb + ((size_t)(((iy0 & 255) << 8) + (ix1 & 255)) << 6);
        const _Float16* hc10 = hb + ((size_t)(((iy1 & 255) << 8) + (ix0 & 255)) << 6);
        const _Float16* hc11 = hb + ((size_t)(((iy1 & 255) << 8) + (ix1 & 255)) << 6);
#pragma unroll
        for (int half = 0; half < 2; ++half) {
            int c0 = (half << 5) + (g << 3);
            v8h a00 = *(const v8h*)(hc00 + c0), b00 = *(const v8h*)(hc00 + c0 + 16);
            v8h a01 = *(const v8h*)(hc01 + c0), b01 = *(const v8h*)(hc01 + c0 + 16);
            v8h a10 = *(const v8h*)(hc10 + c0), b10 = *(const v8h*)(hc10 + c0 + 16);
            v8h a11 = *(const v8h*)(hc11 + c0), b11 = *(const v8h*)(hc11 + c0 + 16);
            AB a;
            a.h2[0] = a00 * W00 + a01 * W01 + a10 * W10 + a11 * W11;  // v_pk_fma_f16
            a.h2[1] = b00 * W00 + b01 * W01 + b10 * W10 + b11 * W11;
            int q = (t << 1) + half;
#pragma unroll
            for (int n = 0; n < 4; ++n) {
                v16h bv = W3f[((q << 2) + n) * 32 + lane];
                acc[n] = __builtin_amdgcn_wmma_f32_16x16x32_f16(
                    false, a.v, false, bv, (short)0, acc[n], false, false);
            }
        }
    }
#pragma unroll
    for (int n = 0; n < 4; ++n) {
        int o = (n << 4) + lc;
        float bo = bdef[o];
        float* op = out + ((size_t)b * 64 + o) * HW + (y << 8) + xBase + (g << 3);
        float4 v0, v1;
        v0.x = fmaxf(acc[n][0] + bo, 0.f); v0.y = fmaxf(acc[n][1] + bo, 0.f);
        v0.z = fmaxf(acc[n][2] + bo, 0.f); v0.w = fmaxf(acc[n][3] + bo, 0.f);
        v1.x = fmaxf(acc[n][4] + bo, 0.f); v1.y = fmaxf(acc[n][5] + bo, 0.f);
        v1.z = fmaxf(acc[n][6] + bo, 0.f); v1.w = fmaxf(acc[n][7] + bo, 0.f);
        *(float4*)(op)     = v0;
        *(float4*)(op + 4) = v1;
    }
}

// ---------------------------------------------------------------------------
extern "C" void kernel_launch(void* const* d_in, const int* in_sizes, int n_in,
                              void* d_out, int out_size, void* d_ws, size_t ws_size,
                              hipStream_t stream) {
    const float* x        = (const float*)d_in[0];
    const float* w_deconv = (const float*)d_in[1];
    const float* b_deconv = (const float*)d_in[2];
    const float* w_off    = (const float*)d_in[3];
    const float* b_off    = (const float*)d_in[4];
    const float* w_def    = (const float*)d_in[5];
    const float* b_def    = (const float*)d_in[6];
    float* out = (float*)d_out;

    char* ws = (char*)d_ws;
    _Float16* xh  = (_Float16*)ws;                    // 16,777,216 B (NHWC f16 x)
    _Float16* hh  = (_Float16*)(ws + 16777216);       // 16,777,216 B (NHWC f16 h)
    float*    off = (float*)   (ws + 33554432);       //  9,437,184 B ([b][pix][18] f32)
    _Float16* W1f = (_Float16*)(ws + 42991616);       //     73,728 B
    _Float16* W2f = (_Float16*)(ws + 43065344);       //     36,864 B
    _Float16* W3f = (_Float16*)(ws + 43102208);       //     73,728 B

    pack_x      <<<32768, 256, 0, stream>>>(x, xh);
    prep_weights<<<144,   256, 0, stream>>>(w_deconv, w_off, w_def, W1f, W2f, W3f);
    conv1_wmma  <<<512,   256, 0, stream>>>(xh, (const v16h*)W1f, b_deconv, hh);
    conv2_wmma  <<<512,   256, 0, stream>>>(hh, (const v16h*)W2f, b_off, off);
    deform_wmma <<<1024,  256, 0, stream>>>(hh, off, (const v16h*)W3f, b_def, out);
}
